// PatchedBertSelfAttention_953482740253
// MI455X (gfx1250) — matrix operand order repaired and checked by reading, not compiled, not run on hardware
//
#include <hip/hip_runtime.h>
#include <hip/hip_bf16.h>

// BERT self-attention with blind-spot diagonal + aux loss, fused flash-style.
// B=8, S=1024, D=768, H=12, dh=64. All matrix math via v_wmma_f32_16x16x32_bf16
// (bf16 operands, f32 accumulation). Attention streams keys with a
// double-buffered async global->LDS pipeline (ASYNCcnt) shared by the 4 waves
// of a block, so K/V are fetched once per block and WMMA overlaps the copies.

#define BB   8
#define SS   1024
#define DD   768
#define HH   12
#define DHH  64

typedef __attribute__((ext_vector_type(16))) __bf16 v16bf;
typedef __attribute__((ext_vector_type(8)))  float  v8f;

// ---------------------------------------------------------------------------
// Async global->LDS copy (16 bytes per call), with synchronous fallback.
// Builtin signature (per hipcc diagnostic): (v4i AS1*, v4i AS3*, imm, imm).
// ---------------------------------------------------------------------------
#if __has_builtin(__builtin_amdgcn_global_load_async_to_lds_b128)
#define ASYNC_COPY 1
#else
#define ASYNC_COPY 0
#endif

typedef int v4i __attribute__((vector_size(16)));
typedef __attribute__((address_space(1))) v4i* gp4_t;   // ptr to AS1 int4
typedef __attribute__((address_space(3))) v4i* lp4_t;   // ptr to AS3 int4

__device__ __forceinline__ void cp16(__bf16* lds_dst, const __bf16* g_src)
{
#if ASYNC_COPY
    // Generic LDS address low 32 bits are the LDS byte offset (ISA 10.2).
    __builtin_amdgcn_global_load_async_to_lds_b128(
        (gp4_t)(uintptr_t)g_src,
        (lp4_t)(unsigned int)(uintptr_t)lds_dst,
        0, 0);
#else
    *(uint4*)lds_dst = *(const uint4*)g_src;
#endif
}

__device__ __forceinline__ void wait_async()
{
#if ASYNC_COPY
#if __has_builtin(__builtin_amdgcn_s_wait_asynccnt)
    __builtin_amdgcn_s_wait_asynccnt(0);
#else
    asm volatile("s_wait_asynccnt 0x0" ::: "memory");
#endif
#endif
}

// ---------------------------------------------------------------------------
// Kernel 1: fused QKV projection. X[8192,768] @ W[768,768] + b, output bf16.
// One wave = one 16x16 output tile; 24 k-steps of 32 => 24 WMMAs per wave.
// Q,K stored [B*H][S][64]; V stored transposed [B*H][64][S] so the PV
// B-fragments in the attention kernel are contiguous along k.
// ---------------------------------------------------------------------------
__global__ __launch_bounds__(128)
void qkv_proj_kernel(const float* __restrict__ X,
                     const float* __restrict__ Wq, const float* __restrict__ bq,
                     const float* __restrict__ Wk, const float* __restrict__ bk,
                     const float* __restrict__ Wv, const float* __restrict__ bvec,
                     __bf16* __restrict__ Qws, __bf16* __restrict__ Kws,
                     __bf16* __restrict__ Vtws)
{
    const int proj = blockIdx.y;                 // 0=Q 1=K 2=V
    const float* Wm = (proj == 0) ? Wq : (proj == 1) ? Wk : Wv;
    const float* bm = (proj == 0) ? bq : (proj == 1) ? bk : bvec;

    const int wave = threadIdx.x >> 5;
    const int lane = threadIdx.x & 31;
    const int col  = lane & 15;
    const int lg   = lane >> 4;                  // lane group 0/1

    const int tile = blockIdx.x * 4 + wave;      // 0 .. 512*48-1
    const int mt   = tile / 48;                  // row tile (over 8192 rows)
    const int nt   = tile % 48;                  // col tile (over 768 cols)

    const int m = mt * 16 + col;                 // A-fragment row for this lane
    const int n = nt * 16 + col;                 // B-fragment / output column

    v8f acc = {};
    for (int kk = 0; kk < DD; kk += 32) {
        // A fragment (16x32 bf16): lane row m, k = 16*lg + e
        v16bf a;
        const float* arow = X + (size_t)m * DD + kk + 16 * lg;
        #pragma unroll
        for (int e = 0; e < 8; ++e) {
            a[e]     = (__bf16)arow[e];
            a[8 + e] = (__bf16)arow[8 + e];
        }
        // B fragment (32x16 bf16): n = col, k = 16*lg + e  (column of W)
        v16bf bfr;
        const float* wcol = Wm + (size_t)(kk + 16 * lg) * DD + n;
        #pragma unroll
        for (int e = 0; e < 16; ++e) bfr[e] = (__bf16)wcol[(size_t)e * DD];
        if (kk + 32 < DD) {
            __builtin_prefetch(wcol + (size_t)32 * DD, 0, 0);  // global_prefetch_b8
        }

        acc = __builtin_amdgcn_wmma_f32_16x16x32_bf16(false, a, false, bfr,
                                                      (short)0, acc, false, false);
    }

    const float bias = bm[n];
    const int h  = n / DHH;
    const int dh = n % DHH;
    #pragma unroll
    for (int r = 0; r < 8; ++r) {
        const int mr = mt * 16 + r + 8 * lg;     // C-layout row
        const int b  = mr >> 10;
        const int s  = mr & (SS - 1);
        const __bf16 val = (__bf16)(acc[r] + bias);
        if (proj == 2) {
            Vtws[((size_t)(b * HH + h) * DHH + dh) * SS + s] = val;
        } else {
            __bf16* dst = (proj == 0) ? Qws : Kws;
            dst[((size_t)(b * HH + h) * SS + s) * DHH + dh] = val;
        }
    }
}

// ---------------------------------------------------------------------------
// Kernel 2: flash attention. 4 waves/block share one (b,h); each wave owns a
// 16-query tile. Per 32-key step: async-stage K(32x64) and V^T(64x32) tiles
// into double-buffered LDS (issued one step ahead), 4 WMMAs for S = Q K^T,
// online softmax with blind-spot diagonal + additive mask, P routed through
// per-wave LDS to A-fragment layout, 4 WMMAs for O += P V.
// ---------------------------------------------------------------------------
__global__ __launch_bounds__(128)
void attention_kernel(const __bf16* __restrict__ Qws,
                      const __bf16* __restrict__ Kws,
                      const __bf16* __restrict__ Vtws,
                      const float*  __restrict__ mask,   // [B,1,1,S]
                      float*        __restrict__ out)    // [B,S,768]
{
    __shared__ __bf16 ktile[2][32][DHH];   // [buf][key][dh]   4KB each
    __shared__ __bf16 vtile[2][DHH][32];   // [buf][dh][key]   4KB each
    __shared__ float  pbuf[4][16 * 32];    // per-wave P       8KB

    const int tid  = threadIdx.x;
    const int wave = tid >> 5;
    const int lane = tid & 31;
    const int col  = lane & 15;
    const int lg   = lane >> 4;

    const int bh = blockIdx.y;                   // 0..95
    const int b  = bh / HH;
    const int h  = bh % HH;
    const int qb = (blockIdx.x * 4 + wave) * 16; // query tile base

    const __bf16* Qb = Qws  + (size_t)bh * SS * DHH;
    const __bf16* Kb = Kws  + (size_t)bh * SS * DHH;
    const __bf16* Vb = Vtws + (size_t)bh * DHH * SS;
    const float*  mr = mask + (size_t)b * SS;

    // Q A-fragments, dh 0..31 and 32..63 (resident for the whole sweep)
    v16bf qa0, qa1;
    {
        const __bf16* qrow = Qb + (size_t)(qb + col) * DHH;
        #pragma unroll
        for (int e = 0; e < 8; ++e) {
            qa0[e]     = qrow[16 * lg + e];
            qa0[8 + e] = qrow[16 * lg + 8 + e];
            qa1[e]     = qrow[32 + 16 * lg + e];
            qa1[8 + e] = qrow[32 + 16 * lg + 8 + e];
        }
    }

    v8f o0 = {}, o1 = {}, o2 = {}, o3 = {};
    float mrun[8], lrun[8];
    #pragma unroll
    for (int r = 0; r < 8; ++r) { mrun[r] = -1e30f; lrun[r] = 0.0f; }

    const float scale = 0.125f;                  // 1/sqrt(64)
    float* myP = pbuf[wave];

    // Cooperative tile stage: 256 16B chunks for K, 256 for V; 2 each/thread.
    auto issue_tile = [&](int kbs, int buf) {
        #pragma unroll
        for (int j = 0; j < 2; ++j) {
            const int c  = tid + 128 * j;              // 0..255
            const int kr = c >> 3, ko = (c & 7) * 8;   // key row, dh offset
            cp16(&ktile[buf][kr][ko], Kb + (size_t)(kbs + kr) * DHH + ko);
            const int vr = c >> 2, vo = (c & 3) * 8;   // dh row, key offset
            cp16(&vtile[buf][vr][vo], Vb + (size_t)vr * SS + kbs + vo);
        }
    };

    issue_tile(0, 0);                             // prime the pipeline

    for (int kbs = 0; kbs < SS; kbs += 32) {
        const int p = (kbs >> 5) & 1;
        wait_async();                             // my async chunks landed
        __syncthreads();                          // everyone's chunks landed
        if (kbs + 32 < SS) issue_tile(kbs + 32, p ^ 1);   // overlap next tile

        // ---- S = Q K^T for 32 keys (two 16-key column blocks) ----
        v8f s0 = {}, s1 = {};
        {
            v16bf k00, k01, k10, k11;
            #pragma unroll
            for (int e = 0; e < 16; ++e) {
                k00[e] = ktile[p][col][16 * lg + e];            // dh 0..31
                k01[e] = ktile[p][col][32 + 16 * lg + e];       // dh 32..63
                k10[e] = ktile[p][16 + col][16 * lg + e];
                k11[e] = ktile[p][16 + col][32 + 16 * lg + e];
            }
            s0 = __builtin_amdgcn_wmma_f32_16x16x32_bf16(false, qa0, false, k00, (short)0, s0, false, false);
            s0 = __builtin_amdgcn_wmma_f32_16x16x32_bf16(false, qa1, false, k01, (short)0, s0, false, false);
            s1 = __builtin_amdgcn_wmma_f32_16x16x32_bf16(false, qa0, false, k10, (short)0, s1, false, false);
            s1 = __builtin_amdgcn_wmma_f32_16x16x32_bf16(false, qa1, false, k11, (short)0, s1, false, false);
        }

        // ---- blind-spot diag, scale, mask, online softmax ----
        const float mv0 = mr[kbs + col];
        const float mv1 = mr[kbs + 16 + col];
        float p0[8], p1[8], nm[8];
        #pragma unroll
        for (int r = 0; r < 8; ++r) {
            const int mg = qb + r + 8 * lg;      // global query row
            float v0 = (kbs + col == mg)      ? mv0 : s0[r] * scale + mv0;
            float v1 = (kbs + 16 + col == mg) ? mv1 : s1[r] * scale + mv1;
            p0[r] = v0; p1[r] = v1;
            float t = fmaxf(v0, v1);
            t = fmaxf(t, __shfl_xor(t, 1));
            t = fmaxf(t, __shfl_xor(t, 2));
            t = fmaxf(t, __shfl_xor(t, 4));
            t = fmaxf(t, __shfl_xor(t, 8));
            nm[r] = fmaxf(mrun[r], t);
        }
        #pragma unroll
        for (int r = 0; r < 8; ++r) {
            const float e0 = __expf(p0[r] - nm[r]);
            const float e1 = __expf(p1[r] - nm[r]);
            p0[r] = e0; p1[r] = e1;
            float rs = e0 + e1;
            rs += __shfl_xor(rs, 1);
            rs += __shfl_xor(rs, 2);
            rs += __shfl_xor(rs, 4);
            rs += __shfl_xor(rs, 8);
            const float alpha = __expf(mrun[r] - nm[r]);
            lrun[r] = lrun[r] * alpha + rs;
            mrun[r] = nm[r];
            o0[r] *= alpha; o1[r] *= alpha; o2[r] *= alpha; o3[r] *= alpha;
        }

        // ---- P (C-layout) -> LDS -> A-fragment (bf16) ----
        #pragma unroll
        for (int r = 0; r < 8; ++r) {
            myP[(r + 8 * lg) * 32 + col]      = p0[r];
            myP[(r + 8 * lg) * 32 + col + 16] = p1[r];
        }
        __syncthreads();
        v16bf pa;
        #pragma unroll
        for (int e = 0; e < 8; ++e) {
            pa[e]     = (__bf16)myP[col * 32 + 16 * lg + e];
            pa[8 + e] = (__bf16)myP[col * 32 + 16 * lg + 8 + e];
        }

        // ---- O += P @ V (V^T tile in LDS makes k contiguous) ----
        #pragma unroll
        for (int f = 0; f < 4; ++f) {
            v16bf vfr;
            #pragma unroll
            for (int e = 0; e < 16; ++e) vfr[e] = vtile[p][16 * f + col][16 * lg + e];
            v8f* op = (f == 0) ? &o0 : (f == 1) ? &o1 : (f == 2) ? &o2 : &o3;
            *op = __builtin_amdgcn_wmma_f32_16x16x32_bf16(false, pa, false, vfr,
                                                          (short)0, *op, false, false);
        }
        // Next iteration's top __syncthreads() protects tile buffers: the
        // overwrite of buffer p happens only after every wave passed it.
    }

    // ---- normalize and store context [B,S,768] ----
    #pragma unroll
    for (int r = 0; r < 8; ++r) {
        const float inv = 1.0f / lrun[r];
        o0[r] *= inv; o1[r] *= inv; o2[r] *= inv; o3[r] *= inv;
    }
    #pragma unroll
    for (int r = 0; r < 8; ++r) {
        const size_t row = (size_t)b * SS + qb + r + 8 * lg;
        float* orow = out + row * DD + h * DHH;
        orow[col]      = o0[r];
        orow[16 + col] = o1[r];
        orow[32 + col] = o2[r];
        orow[48 + col] = o3[r];
    }
}

// ---------------------------------------------------------------------------
// Kernel 3: aux loss. Diagonal of (blind-spotted, scaled, masked) scores is
// exactly mask[b,0,0,q], so loss = LMBDA * H * sum(mask).
// ---------------------------------------------------------------------------
__global__ __launch_bounds__(256)
void loss_kernel(const float* __restrict__ mask, float* __restrict__ lossOut)
{
    __shared__ float red[256];
    float s = 0.0f;
    for (int i = threadIdx.x; i < BB * SS; i += 256) s += mask[i];
    red[threadIdx.x] = s;
    __syncthreads();
    for (int w = 128; w > 0; w >>= 1) {
        if ((int)threadIdx.x < w) red[threadIdx.x] += red[threadIdx.x + w];
        __syncthreads();
    }
    if (threadIdx.x == 0) lossOut[0] = 0.01f * (float)HH * red[0];
}

// ---------------------------------------------------------------------------
extern "C" void kernel_launch(void* const* d_in, const int* in_sizes, int n_in,
                              void* d_out, int out_size, void* d_ws, size_t ws_size,
                              hipStream_t stream)
{
    const float* hs   = (const float*)d_in[0];
    const float* mask = (const float*)d_in[1];
    const float* Wq   = (const float*)d_in[2];
    const float* bq   = (const float*)d_in[3];
    const float* Wk   = (const float*)d_in[4];
    const float* bk   = (const float*)d_in[5];
    const float* Wv   = (const float*)d_in[6];
    const float* bvp  = (const float*)d_in[7];

    float* ctx  = (float*)d_out;
    float* loss = ctx + (size_t)BB * SS * DD;

    const size_t elems = (size_t)BB * HH * SS * DHH;   // 6,291,456 per tensor
    __bf16* Qws = (__bf16*)d_ws;
    __bf16* Kws = Qws + elems;
    __bf16* Vt  = Kws + elems;

    // 512 row-tiles * 48 col-tiles = 24576 tiles; 4 waves/block -> 6144 blocks,
    // grid.y selects Q/K/V projection.
    qkv_proj_kernel<<<dim3(6144, 3, 1), 128, 0, stream>>>(
        hs, Wq, bq, Wk, bk, Wv, bvp, Qws, Kws, Vt);

    // 96 (b,h) pairs * 64 query tiles; 4 waves/block -> grid (16, 96).
    attention_kernel<<<dim3(16, 96, 1), 128, 0, stream>>>(Qws, Kws, Vt, mask, ctx);

    loss_kernel<<<1, 256, 0, stream>>>(mask, loss);
}
